// SAN_81844896793371
// MI455X (gfx1250) — matrix operand order repaired and checked by reading, not compiled, not run on hardware
//
#include <hip/hip_runtime.h>

// ---------------------------------------------------------------------------
// Types for CDNA5 WMMA (wave32): D(f32 16x16) = A(bf16 16x32) x B(bf16 32x16) + C
// ---------------------------------------------------------------------------
typedef __attribute__((ext_vector_type(16))) __bf16 v16bf;
typedef __attribute__((ext_vector_type(8)))  __bf16 v8bf;
typedef __attribute__((ext_vector_type(8)))  float  v8f;

#define HID 128
#define NGRAPH 1024
#define GOAL_D 20
#define EMB_D 1280
#define POL_D 5

// ---------------------------------------------------------------------------
// Generic fill
// ---------------------------------------------------------------------------
__global__ void gcn_fill_kernel(float* __restrict__ p, float v, long n) {
  long i = (long)blockIdx.x * blockDim.x + threadIdx.x;
  if (i < n) p[i] = v;
}

// deg[col[e]] += 1  (deg pre-filled with 1.0 for the self loop)
__global__ void gcn_deg_kernel(const int* __restrict__ ei, float* __restrict__ deg, int E) {
  int e = blockIdx.x * blockDim.x + threadIdx.x;
  if (e < E) atomicAdd(&deg[ei[E + e]], 1.0f);
}

// dis[i] = rsqrt(deg[i])   (deg >= 1 always)
__global__ void gcn_rsqrt_kernel(float* __restrict__ d, int N) {
  int i = blockIdx.x * blockDim.x + threadIdx.x;
  if (i < N) d[i] = rsqrtf(d[i]);
}

// ---------------------------------------------------------------------------
// WMMA GEMM: H[N x 128] = X[N x K] @ W[K x 128], K compile-time in {16,128}.
// Block = 256 threads = 8 waves; wave w owns rows [blk*128 + 16w, +16).
// Each wave holds 8 f32 16x16 accumulators (full 128 output channels).
// W staged in LDS as bf16, transposed [col][k], zero-padded to KP=max(K,32),
// stride KP+8 (16B pad) so per-lane column reads cover all 64 LDS banks.
// Epilogue: single wave-uniform scalar branch; guarded stores only in the
// one partial tail block.
// ---------------------------------------------------------------------------
template <int K>
__global__ __launch_bounds__(256) void gcn_gemm_wmma_kernel(
    const float* __restrict__ X, const float* __restrict__ W,
    float* __restrict__ H, int N) {
  constexpr int KP = (K < 32) ? 32 : K;   // padded K (multiple of WMMA step)
  constexpr int LSTR = KP + 8;            // LDS stride in elements (+16B pad)
  __shared__ __bf16 Wt[HID * LSTR];

  for (int i = threadIdx.x; i < HID * KP; i += 256) {
    int c = i / KP, k = i - c * KP;
    Wt[c * LSTR + k] = (k < K) ? (__bf16)W[k * HID + c] : (__bf16)0.0f;
  }
  __syncthreads();

  const int lane = threadIdx.x & 31;
  const int wave = threadIdx.x >> 5;
  const int half = lane >> 4;   // 0: lanes 0-15, 1: lanes 16-31
  const int m    = lane & 15;   // row within tile (A) / col within tile (B)
  const int baseRow = blockIdx.x * 128 + wave * 16;

  v8f acc[8] = {};

  int aRow = baseRow + m;
  if (aRow >= N) aRow = N - 1;             // clamp (stores are guarded)
  const float* xr = X + (long)aRow * K;
  const int kA = half * 8;                 // A-fragment lane K base (0 or 8)
  const int kB = half * 8;                 // B-fragment lane K base (0 or 8)

#pragma unroll
  for (int k0 = 0; k0 < KP; k0 += 32) {
    // A fragment: group g covers K = k0 + 16g + kA + [0,8); since K is a
    // multiple of 16 and kA<=8, validity is decided at compile time.
    v16bf a;
#pragma unroll
    for (int g = 0; g < 2; ++g) {
      if (k0 + g * 16 < K) {
        const float* ap = xr + k0 + g * 16 + kA;
        float4 f0 = *(const float4*)(ap);
        float4 f1 = *(const float4*)(ap + 4);
        a[g * 8 + 0] = (__bf16)f0.x; a[g * 8 + 1] = (__bf16)f0.y;
        a[g * 8 + 2] = (__bf16)f0.z; a[g * 8 + 3] = (__bf16)f0.w;
        a[g * 8 + 4] = (__bf16)f1.x; a[g * 8 + 5] = (__bf16)f1.y;
        a[g * 8 + 6] = (__bf16)f1.z; a[g * 8 + 7] = (__bf16)f1.w;
      } else {
#pragma unroll
        for (int e = 0; e < 8; ++e) a[g * 8 + e] = (__bf16)0.0f;
      }
    }
#pragma unroll
    for (int j = 0; j < 8; ++j) {          // 8 column tiles -> 128 channels
      const __bf16* wc = &Wt[(j * 16 + m) * LSTR + k0 + kB];
      v8bf lo = *(const v8bf*)(wc);
      v8bf hi = *(const v8bf*)(wc + 16);
      v16bf b = __builtin_shufflevector(lo, hi, 0, 1, 2, 3, 4, 5, 6, 7,
                                        8, 9, 10, 11, 12, 13, 14, 15);
      acc[j] = __builtin_amdgcn_wmma_f32_16x16x32_bf16(
          false, a, false, b, (short)0, acc[j], false, false);
    }
  }

  // D layout: VGPR r, lanes 0-15 -> row r, lanes 16-31 -> row 8+r; col = lane&15
  // One base pointer per lane; r/j offsets are compile-time immediates.
  float* hb = H + (long)(baseRow + half * 8) * HID + m;
  if (baseRow + 16 <= N) {                 // wave-uniform: all 16 rows valid
#pragma unroll
    for (int j = 0; j < 8; ++j)
#pragma unroll
      for (int r = 0; r < 8; ++r)
        hb[r * HID + j * 16] = acc[j][r];
  } else {                                 // only the final partial block
#pragma unroll
    for (int j = 0; j < 8; ++j)
#pragma unroll
      for (int r = 0; r < 8; ++r)
        if (baseRow + half * 8 + r < N) hb[r * HID + j * 16] = acc[j][r];
  }
}

// ---------------------------------------------------------------------------
// Edge scatter: one wave per edge; lane handles 4 of 128 channels (float4).
// Out[col] += H[row] * dis[row]*dis[col]   via global_atomic_add_f32
// ---------------------------------------------------------------------------
__global__ __launch_bounds__(256) void gcn_scatter_kernel(
    const float* __restrict__ H, const int* __restrict__ ei,
    const float* __restrict__ dis, float* __restrict__ Out, int E) {
  int lane = threadIdx.x & 31;
  int e = (int)(((long)blockIdx.x * blockDim.x + threadIdx.x) >> 5);
  if (e >= E) return;
  int r = ei[e], c = ei[E + e];
  float nrm = dis[r] * dis[c];
  float4 v = ((const float4*)(H + (long)r * HID))[lane];
  float* o = Out + (long)c * HID + lane * 4;
  atomicAdd(o + 0, v.x * nrm);
  atomicAdd(o + 1, v.y * nrm);
  atomicAdd(o + 2, v.z * nrm);
  atomicAdd(o + 3, v.w * nrm);
}

// agg = relu(agg + lin*dis^2 + bias)   (self-loop folded in analytically)
__global__ void gcn_finalize_kernel(float* __restrict__ agg, const float* __restrict__ lin,
                                    const float* __restrict__ dis, const float* __restrict__ bias,
                                    int N) {
  long i = (long)blockIdx.x * blockDim.x + threadIdx.x;
  if (i >= (long)N * HID) return;
  int node = (int)(i >> 7), c = (int)(i & 127);
  float d = dis[node];
  float v = agg[i] + lin[i] * d * d + bias[c];
  agg[i] = fmaxf(v, 0.0f);
}

// one wave per node: pooled[batch[i]] += h[i]; cnt[batch[i]] += 1
__global__ __launch_bounds__(256) void gcn_pool_kernel(
    const float* __restrict__ H, const int* __restrict__ batch,
    float* __restrict__ pooled, float* __restrict__ cnt, int N) {
  int lane = threadIdx.x & 31;
  int i = (int)(((long)blockIdx.x * blockDim.x + threadIdx.x) >> 5);
  if (i >= N) return;
  int g = batch[i];
  float4 v = ((const float4*)(H + (long)i * HID))[lane];
  float* p = pooled + (long)g * HID + lane * 4;
  atomicAdd(p + 0, v.x);
  atomicAdd(p + 1, v.y);
  atomicAdd(p + 2, v.z);
  atomicAdd(p + 3, v.w);
  if (lane == 0) atomicAdd(&cnt[g], 1.0f);
}

// per-graph heads: block g, 128 threads, mean row cached in LDS
__global__ __launch_bounds__(128) void gcn_heads_kernel(
    const float* __restrict__ pooled, const float* __restrict__ cnt,
    const float* __restrict__ Wg, const float* __restrict__ bg,
    const float* __restrict__ We, const float* __restrict__ be,
    const float* __restrict__ Wp, const float* __restrict__ bp,
    float* __restrict__ out) {
  __shared__ float mean[HID];
  int g = blockIdx.x, t = threadIdx.x;
  float c = fmaxf(cnt[g], 1.0f);
  mean[t] = pooled[(long)g * HID + t] / c;
  __syncthreads();

  const long goalOff = 0;
  const long embOff  = (long)NGRAPH * GOAL_D;
  const long polOff  = embOff + (long)NGRAPH * EMB_D;

  if (t < GOAL_D) {
    float s = bg[t];
#pragma unroll 8
    for (int k = 0; k < HID; ++k) s += mean[k] * Wg[k * GOAL_D + t];
    out[goalOff + (long)g * GOAL_D + t] = 1.0f / (1.0f + __expf(-s));
  }
#pragma unroll
  for (int jj = 0; jj < EMB_D / HID; ++jj) {   // 10 outputs per thread
    int j = t * (EMB_D / HID) + jj;
    float s = be[j];
#pragma unroll 8
    for (int k = 0; k < HID; ++k) s += mean[k] * We[k * EMB_D + j];
    out[embOff + (long)g * EMB_D + j] = s;
  }
  if (t < POL_D) {
    float s = bp[t];
#pragma unroll 8
    for (int k = 0; k < HID; ++k) s += mean[k] * Wp[k * POL_D + t];
    out[polOff + (long)g * POL_D + t] = s;
  }
}

// ---------------------------------------------------------------------------
extern "C" void kernel_launch(void* const* d_in, const int* in_sizes, int n_in,
                              void* d_out, int out_size, void* d_ws, size_t ws_size,
                              hipStream_t stream) {
  const float* x     = (const float*)d_in[0];
  const int*   ei    = (const int*)d_in[1];     // [2, E]
  const int*   batch = (const int*)d_in[2];
  const float* W1 = (const float*)d_in[3];
  const float* b1 = (const float*)d_in[4];
  const float* W2 = (const float*)d_in[5];
  const float* b2 = (const float*)d_in[6];
  const float* Wg = (const float*)d_in[7];
  const float* bg = (const float*)d_in[8];
  const float* We = (const float*)d_in[9];
  const float* be = (const float*)d_in[10];
  const float* Wp = (const float*)d_in[11];
  const float* bp = (const float*)d_in[12];
  float* out = (float*)d_out;

  const int N = in_sizes[0] / 16;   // 100000
  const int E = in_sizes[1] / 2;    // 1600000

  // workspace layout (floats)
  float* ws = (float*)d_ws;
  size_t off = ((size_t)N + 1023) & ~(size_t)1023;
  float* dis    = ws;                            // N (deg -> rsqrt in place)
  float* bufA   = ws + off;                      // N*128
  float* bufB   = bufA + (size_t)N * HID;        // N*128
  float* pooled = bufB + (size_t)N * HID;        // 1024*128
  float* cnt    = pooled + (size_t)NGRAPH * HID; // 1024 (contiguous with pooled)

  auto cdiv = [](long a, long b) { return (int)((a + b - 1) / b); };
  const long NH = (long)N * HID;
  const int gemmBlocks = cdiv(N, 128);
  const int scatBlocks = cdiv((long)E * 32, 256);
  const int poolBlocks = cdiv((long)N * 32, 256);

  // --- degree / symmetric norm ---
  gcn_fill_kernel<<<cdiv(N, 256), 256, 0, stream>>>(dis, 1.0f, N);
  gcn_deg_kernel<<<cdiv(E, 256), 256, 0, stream>>>(ei, dis, E);
  gcn_rsqrt_kernel<<<cdiv(N, 256), 256, 0, stream>>>(dis, N);

  // --- layer 1: lin (WMMA, K=16) -> scatter -> self-loop+bias+relu ---
  gcn_gemm_wmma_kernel<16><<<gemmBlocks, 256, 0, stream>>>(x, W1, bufA, N);
  gcn_fill_kernel<<<cdiv(NH, 256), 256, 0, stream>>>(bufB, 0.0f, NH);
  gcn_scatter_kernel<<<scatBlocks, 256, 0, stream>>>(bufA, ei, dis, bufB, E);
  gcn_finalize_kernel<<<cdiv(NH, 256), 256, 0, stream>>>(bufB, bufA, dis, b1, N);

  // --- layer 2 (WMMA, K=128) ---
  gcn_gemm_wmma_kernel<128><<<gemmBlocks, 256, 0, stream>>>(bufB, W2, bufA, N);
  gcn_fill_kernel<<<cdiv(NH, 256), 256, 0, stream>>>(bufB, 0.0f, NH);
  gcn_scatter_kernel<<<scatBlocks, 256, 0, stream>>>(bufA, ei, dis, bufB, E);
  gcn_finalize_kernel<<<cdiv(NH, 256), 256, 0, stream>>>(bufB, bufA, dis, b2, N);

  // --- mean pool + heads ---
  gcn_fill_kernel<<<cdiv((long)NGRAPH * HID + NGRAPH, 256), 256, 0, stream>>>(
      pooled, 0.0f, (long)NGRAPH * HID + NGRAPH);
  gcn_pool_kernel<<<poolBlocks, 256, 0, stream>>>(bufB, batch, pooled, cnt, N);
  gcn_heads_kernel<<<NGRAPH, HID, 0, stream>>>(pooled, cnt, Wg, bg, We, be, Wp, bp, out);
}
